// SelfAttention_v2_90855738180063
// MI455X (gfx1250) — matrix operand order repaired and checked by reading, not compiled, not run on hardware
//
#include <hip/hip_runtime.h>

#define SEQ  8192
#define DIN  1024
#define DOUT 1024

typedef __bf16 bf16;
typedef __attribute__((ext_vector_type(16))) __bf16 v16bf;
typedef __attribute__((ext_vector_type(8)))  __bf16 v8bf;
typedef __attribute__((ext_vector_type(4)))  __bf16 v4bf;
typedef __attribute__((ext_vector_type(8)))  float  v8f;
typedef __attribute__((ext_vector_type(4)))  float  v4f;

union FragU { v16bf v; v8bf h[2]; };

static __device__ __forceinline__ v8f wmma_bf16(v16bf a, v16bf b, v8f c) {
  // (neg_a, A, neg_b, B, c_mod, C, reuse_a, reuse_b)
  return __builtin_amdgcn_wmma_f32_16x16x32_bf16(false, a, false, b, (short)0, c,
                                                 false, false);
}

static __device__ __forceinline__ v8f zero8() {
  v8f z = {0.f, 0.f, 0.f, 0.f, 0.f, 0.f, 0.f, 0.f};
  return z;
}

// A fragment (16x32 bf16). Lane = hi*16 + ln, holds row M = ln.
// VGPRs 0..3 = K[k0 + hi*8 .. +7], VGPRs 4..7 = K[k0 + 16 + hi*8 .. +7]
static __device__ __forceinline__ v16bf load_a(const bf16* row, int k0, int hi) {
  FragU f;
  f.h[0] = *(const v8bf*)(row + k0 + hi * 8);
  f.h[1] = *(const v8bf*)(row + k0 + 16 + hi * 8);
  return f.v;
}

// B fragment (32x16 bf16). Lane holds column N = ln; K[k0 + hi*8 .. +7], K[k0 + 16 + hi*8 .. +7]
static __device__ __forceinline__ v16bf load_b(const bf16* col, int k0, int hi) {
  FragU f;
  f.h[0] = *(const v8bf*)(col + k0 + hi * 8);
  f.h[1] = *(const v8bf*)(col + k0 + 16 + hi * 8);
  return f.v;
}

// ---------------------------------------------------------------- convert X
__global__ void __launch_bounds__(256) k_cvt_bf16(const float* __restrict__ in,
                                                  bf16* __restrict__ out) {
  size_t i = ((size_t)blockIdx.x * 256 + threadIdx.x) * 4;
  v4f v = *(const v4f*)(in + i);
  v4bf o;
#pragma unroll
  for (int j = 0; j < 4; ++j) o[j] = (bf16)v[j];
  *(v4bf*)(out + i) = o;
}

// ------------------------------------------------- transpose W -> bf16 Wt
// W: [DIN, DOUT] f32 row-major ->  Wt: [DOUT, DIN] bf16 row-major
__global__ void __launch_bounds__(256) k_transpose_w(const float* __restrict__ W,
                                                     bf16* __restrict__ Wt) {
  __shared__ float tile[32][33];
  const int tx = threadIdx.x & 31;
  const int ty = threadIdx.x >> 5;  // 0..7
#pragma unroll
  for (int i = 0; i < 4; ++i) {
    int k = blockIdx.x * 32 + ty + i * 8;
    int n = blockIdx.y * 32 + tx;
    tile[ty + i * 8][tx] = W[(size_t)k * DOUT + n];
  }
  __syncthreads();
#pragma unroll
  for (int i = 0; i < 4; ++i) {
    int n = blockIdx.y * 32 + ty + i * 8;
    int k = blockIdx.x * 32 + tx;
    Wt[(size_t)n * DIN + k] = (bf16)tile[tx][ty + i * 8];
  }
}

// --------------------------------------------------------- fused QKV GEMM
// Wave tile 32x64 (2 m-tiles x 4 n-tiles): A frags reused 4x, B frags 2x.
// WG = 8 waves (2 along M, 4 along N) -> 64 x 256 tile.
// z==0 -> Qb (row-major, pre-scaled by 1/sqrt(d)); z==1 -> Kb; z==2 -> Vt^T
__global__ void __launch_bounds__(256) k_gemm_qkv(
    const bf16* __restrict__ X, const bf16* __restrict__ Wtq,
    const bf16* __restrict__ Wtk, const bf16* __restrict__ Wtv,
    bf16* __restrict__ Qb, bf16* __restrict__ Kb, bf16* __restrict__ Vt) {
  const int z    = blockIdx.z;
  const bf16* Wt = (z == 0) ? Wtq : ((z == 1) ? Wtk : Wtv);
  const int tid  = threadIdx.x;
  const int wave = tid >> 5;
  const int lane = tid & 31;
  const int ln   = lane & 15;
  const int hi   = lane >> 4;
  const int mbase = blockIdx.x * 64 + (wave & 1) * 32;
  const int nbase = blockIdx.y * 256 + (wave >> 1) * 64;

  v8f acc[2][4];
#pragma unroll
  for (int i = 0; i < 2; ++i)
#pragma unroll
    for (int j = 0; j < 4; ++j) acc[i][j] = zero8();

  const bf16* Xr[2];
  const bf16* Wr[4];
#pragma unroll
  for (int mi = 0; mi < 2; ++mi) Xr[mi] = X + (size_t)(mbase + mi * 16 + ln) * DIN;
#pragma unroll
  for (int ni = 0; ni < 4; ++ni) Wr[ni] = Wt + (size_t)(nbase + ni * 16 + ln) * DIN;

  for (int kk = 0; kk < DIN; kk += 32) {
    v16bf a[2], b[4];
#pragma unroll
    for (int mi = 0; mi < 2; ++mi) a[mi] = load_a(Xr[mi], kk, hi);
#pragma unroll
    for (int ni = 0; ni < 4; ++ni) b[ni] = load_b(Wr[ni], kk, hi);
#pragma unroll
    for (int mi = 0; mi < 2; ++mi)
#pragma unroll
      for (int ni = 0; ni < 4; ++ni) acc[mi][ni] = wmma_bf16(a[mi], b[ni], acc[mi][ni]);
  }

  if (z < 2) {
    bf16* Y           = (z == 0) ? Qb : Kb;
    const float scale = (z == 0) ? 0.03125f : 1.0f;  // 1/sqrt(1024)
#pragma unroll
    for (int mi = 0; mi < 2; ++mi)
#pragma unroll
      for (int ni = 0; ni < 4; ++ni)
#pragma unroll
        for (int r = 0; r < 8; ++r) {
          int m = mbase + mi * 16 + hi * 8 + r;
          int n = nbase + ni * 16 + ln;
          Y[(size_t)m * DOUT + n] = (bf16)(acc[mi][ni][r] * scale);
        }
  } else {
#pragma unroll
    for (int mi = 0; mi < 2; ++mi)
#pragma unroll
      for (int ni = 0; ni < 4; ++ni) {
        v8bf pk;
#pragma unroll
        for (int r = 0; r < 8; ++r) pk[r] = (bf16)acc[mi][ni][r];
        int n  = nbase + ni * 16 + ln;
        int m0 = mbase + mi * 16 + hi * 8;
        *(v8bf*)(Vt + (size_t)n * SEQ + m0) = pk;  // contiguous 16B store
      }
  }
}

// -------------------------------------------------------- flash attention
// One workgroup (8 waves) = 32 query rows; keys processed in blocks of 128.
// Wave w: S tiles (2 row-halves) for keys [kb+16w, kb+16w+16), reusing the
// K B-fragment 2x; owns output columns [128w, 128w+128) for all 32 rows,
// reusing each V B-fragment 2x and each P A-fragment 8x.
__global__ void __launch_bounds__(256) k_flash(const bf16* __restrict__ Qb,
                                               const bf16* __restrict__ Kb,
                                               const bf16* __restrict__ Vt,
                                               float* __restrict__ Out) {
  __shared__ alignas(16) bf16 sQ[32 * DIN];   // 64 KB
  __shared__ alignas(16) bf16 sP[32 * 128];   // 8 KB
  __shared__ float sMax[8][32];
  __shared__ float sSum[8][32];
  __shared__ float sMrun[32];
  __shared__ float sLrun[32];
  __shared__ float sAlpha[32];

  const int tid  = threadIdx.x;
  const int wave = tid >> 5;
  const int lane = tid & 31;
  const int ln   = lane & 15;
  const int hi   = lane >> 4;
  const int r0   = blockIdx.x * 32;   // first query row of this block
  const int nO   = wave * 128;        // this wave's d_out chunk

  {  // stage Q row-block into LDS (32 x 1024 bf16)
    const v8bf* src = (const v8bf*)(Qb + (size_t)r0 * DIN);
    v8bf* dst = (v8bf*)sQ;
#pragma unroll
    for (int i = 0; i < (32 * DIN / 8) / 256; ++i)
      dst[tid + i * 256] = src[tid + i * 256];
  }
  if (tid < 32) { sMrun[tid] = -__builtin_inff(); sLrun[tid] = 0.0f; }
  __syncthreads();

  v8f O0[8], O1[8];
#pragma unroll
  for (int i = 0; i < 8; ++i) { O0[i] = zero8(); O1[i] = zero8(); }

  const bf16* Qr0 = sQ + ln * DIN;
  const bf16* Qr1 = sQ + (16 + ln) * DIN;
  const bf16* Pr0 = sP + ln * 128;
  const bf16* Pr1 = sP + (16 + ln) * 128;

  for (int kb = 0; kb < SEQ; kb += 128) {
    const bf16* Krow = Kb + (size_t)(kb + wave * 16 + ln) * DOUT;
    if (kb + 128 < SEQ) {  // hint next key/value block toward L2/L0
      __builtin_prefetch(Krow + (size_t)128 * DOUT, 0, 3);
      __builtin_prefetch(Vt + (size_t)(nO + ln) * SEQ + kb + 128, 0, 3);
    }

    // ---- S tiles: 2 row-halves vs this wave's 16 keys ----
    v8f s0 = zero8(), s1 = zero8();
    for (int kk = 0; kk < DIN; kk += 32) {
      v16bf b  = load_b(Krow, kk, hi);   // global (L2-resident), reused 2x
      v16bf a0 = load_a(Qr0, kk, hi);    // ds_load_b128 x2
      v16bf a1 = load_a(Qr1, kk, hi);
      s0 = wmma_bf16(a0, b, s0);
      s1 = wmma_bf16(a1, b, s1);
    }

    // ---- local row max across the 16 lanes of each half-wave ----
#pragma unroll
    for (int r = 0; r < 8; ++r) {
      float v0 = s0[r], v1 = s1[r];
      v0 = fmaxf(v0, __shfl_xor(v0, 1)); v1 = fmaxf(v1, __shfl_xor(v1, 1));
      v0 = fmaxf(v0, __shfl_xor(v0, 2)); v1 = fmaxf(v1, __shfl_xor(v1, 2));
      v0 = fmaxf(v0, __shfl_xor(v0, 4)); v1 = fmaxf(v1, __shfl_xor(v1, 4));
      v0 = fmaxf(v0, __shfl_xor(v0, 8)); v1 = fmaxf(v1, __shfl_xor(v1, 8));
      if (ln == 0) {
        sMax[wave][hi * 8 + r]      = v0;
        sMax[wave][16 + hi * 8 + r] = v1;
      }
    }
    __syncthreads();  // (1) also fences previous iteration's P/alpha readers

    if (tid < 32) {
      float mb = sMax[0][tid];
#pragma unroll
      for (int w = 1; w < 8; ++w) mb = fmaxf(mb, sMax[w][tid]);
      float mold  = sMrun[tid];
      float mnew  = fmaxf(mold, mb);
      sMrun[tid]  = mnew;
      sAlpha[tid] = __expf(mold - mnew);  // first block: exp(-inf) = 0
    }
    __syncthreads();  // (2)

    float al0[8], al1[8], rm0[8], rm1[8];
#pragma unroll
    for (int r = 0; r < 8; ++r) {
      al0[r] = sAlpha[hi * 8 + r];      rm0[r] = sMrun[hi * 8 + r];
      al1[r] = sAlpha[16 + hi * 8 + r]; rm1[r] = sMrun[16 + hi * 8 + r];
    }

    // ---- exp, row sums, publish P tile (bf16) ----
#pragma unroll
    for (int r = 0; r < 8; ++r) {
      float p0 = __expf(s0[r] - rm0[r]);
      float p1 = __expf(s1[r] - rm1[r]);
      sP[(hi * 8 + r) * 128 + wave * 16 + ln]        = (bf16)p0;
      sP[(16 + hi * 8 + r) * 128 + wave * 16 + ln]   = (bf16)p1;
      float t0 = p0, t1 = p1;
      t0 += __shfl_xor(t0, 1); t1 += __shfl_xor(t1, 1);
      t0 += __shfl_xor(t0, 2); t1 += __shfl_xor(t1, 2);
      t0 += __shfl_xor(t0, 4); t1 += __shfl_xor(t1, 4);
      t0 += __shfl_xor(t0, 8); t1 += __shfl_xor(t1, 8);
      if (ln == 0) {
        sSum[wave][hi * 8 + r]      = t0;
        sSum[wave][16 + hi * 8 + r] = t1;
      }
    }
    __syncthreads();  // (3) P + sums visible

    if (tid < 32) {
      float accl = 0.0f;
#pragma unroll
      for (int w = 0; w < 8; ++w) accl += sSum[w][tid];
      sLrun[tid] = sLrun[tid] * sAlpha[tid] + accl;
    }

    // ---- rescale O, accumulate O += P(32x128) x V_block slice ----
#pragma unroll
    for (int nt = 0; nt < 8; ++nt)
#pragma unroll
      for (int r = 0; r < 8; ++r) { O0[nt][r] *= al0[r]; O1[nt][r] *= al1[r]; }

#pragma unroll
    for (int k2 = 0; k2 < 4; ++k2) {
      v16bf aP0 = load_a(Pr0, k2 * 32, hi);
      v16bf aP1 = load_a(Pr1, k2 * 32, hi);
#pragma unroll
      for (int nt = 0; nt < 8; ++nt) {
        const bf16* Vrow = Vt + (size_t)(nO + nt * 16 + ln) * SEQ + kb;
        v16bf bV = load_b(Vrow, k2 * 32, hi);   // reused for both row-halves
        O0[nt] = wmma_bf16(aP0, bV, O0[nt]);
        O1[nt] = wmma_bf16(aP1, bV, O1[nt]);
      }
    }
  }
  __syncthreads();

  float li0[8], li1[8];
#pragma unroll
  for (int r = 0; r < 8; ++r) {
    li0[r] = 1.0f / sLrun[hi * 8 + r];
    li1[r] = 1.0f / sLrun[16 + hi * 8 + r];
  }
#pragma unroll
  for (int nt = 0; nt < 8; ++nt)
#pragma unroll
    for (int r = 0; r < 8; ++r) {
      int col = nO + nt * 16 + ln;
      int m0  = r0 + hi * 8 + r;
      int m1  = r0 + 16 + hi * 8 + r;
      Out[(size_t)m0 * DOUT + col] = O0[nt][r] * li0[r];
      Out[(size_t)m1 * DOUT + col] = O1[nt][r] * li1[r];
    }
}

// ------------------------------------------------------------------ launch
extern "C" void kernel_launch(void* const* d_in, const int* in_sizes, int n_in,
                              void* d_out, int out_size, void* d_ws, size_t ws_size,
                              hipStream_t stream) {
  const float* x  = (const float*)d_in[0];
  const float* Wq = (const float*)d_in[1];
  const float* Wk = (const float*)d_in[2];
  const float* Wv = (const float*)d_in[3];
  float* out = (float*)d_out;

  char* p  = (char*)d_ws;
  bf16* Xb  = (bf16*)p; p += (size_t)SEQ * DIN  * sizeof(bf16);
  bf16* Wtq = (bf16*)p; p += (size_t)DIN * DOUT * sizeof(bf16);
  bf16* Wtk = (bf16*)p; p += (size_t)DIN * DOUT * sizeof(bf16);
  bf16* Wtv = (bf16*)p; p += (size_t)DIN * DOUT * sizeof(bf16);
  bf16* Qb  = (bf16*)p; p += (size_t)SEQ * DOUT * sizeof(bf16);
  bf16* Kb  = (bf16*)p; p += (size_t)SEQ * DOUT * sizeof(bf16);
  bf16* Vt  = (bf16*)p; p += (size_t)SEQ * DOUT * sizeof(bf16);

  k_cvt_bf16<<<(SEQ * DIN) / 1024, 256, 0, stream>>>(x, Xb);

  dim3 gt(DIN / 32, DOUT / 32);
  k_transpose_w<<<gt, 256, 0, stream>>>(Wq, Wtq);
  k_transpose_w<<<gt, 256, 0, stream>>>(Wk, Wtk);
  k_transpose_w<<<gt, 256, 0, stream>>>(Wv, Wtv);

  k_gemm_qkv<<<dim3(SEQ / 64, DOUT / 256, 3), 256, 0, stream>>>(Xb, Wtq, Wtk, Wtv,
                                                                Qb, Kb, Vt);

  k_flash<<<SEQ / 32, 256, 0, stream>>>(Qb, Kb, Vt, out);
}
